// RegionProposalNetwork_84688165143177
// MI455X (gfx1250) — compile-verified, matrix operand order checked by reading, not run on hardware
//
#include <hip/hip_runtime.h>
#include <hip/hip_fp16.h>

// ---------------------------------------------------------------------------
// RPN forward for MI455X (gfx1250, wave32, WMMA + async/TDM data movement).
// conv3x3 (256->256) as implicit GEMM with v_wmma_f32_16x16x32_f16,
// A tiles: global_load_async_to_lds_b128 (saddr form, ASYNCcnt),
// B tiles: Tensor Data Mover (tensor_load_to_lds, TENSORcnt),
// double-buffered LDS (K=64 per stage, 8 WMMA per barrier), then 1x1 heads
// GEMM, decode / top-k / NMS.
// ---------------------------------------------------------------------------

typedef __attribute__((ext_vector_type(16))) _Float16 v16h;
typedef __attribute__((ext_vector_type(8)))  float    v8f;
typedef __attribute__((ext_vector_type(4)))  unsigned u32x4;
typedef __attribute__((ext_vector_type(8)))  unsigned u32x8;

#define B_IMG   8
#define HW      4096            // 64*64
#define M_TOT   32768           // 8*64*64
#define C_IN    256
#define K3      2304            // 256*9
#define N_HEAD  64              // 12 obj + 48 delta + 4 pad
#define ANCH    12
#define N_ANCH  49152           // 4096*12 per image
#define TOPK    1000
#define KEEP    300
#define LDSH    72              // LDS row stride in halves (144B, 16B aligned)

// ---------------------------------------------------------------------------
// CDNA5 async / TDM helpers
// ---------------------------------------------------------------------------
__device__ __forceinline__ unsigned lds_off(const void* p) {
    return (unsigned)(size_t)p;   // low 32 bits of shared-aperture flat addr
}

// Per-lane async copy, GVS form: 16B global -> LDS, tracked by ASYNCcnt.
__device__ __forceinline__ void async_g2l_b128_s(unsigned ldsByteOff, unsigned voffBytes,
                                                 unsigned long long saddr) {
    asm volatile("global_load_async_to_lds_b128 %0, %1, %2"
                 :: "v"(ldsByteOff), "v"(voffBytes), "s"(saddr)
                 : "memory");
}
__device__ __forceinline__ void wait_async0() {
    asm volatile("s_wait_asynccnt 0x0" ::: "memory");
}

// TDM: load a 2D tile (64 elems x 64 rows, f16) of a row-major matrix into
// LDS with a 144B row stride (pad 16B after every 128B). One wave issues.
__device__ __forceinline__ void tdm_load_tile_64x64_f16(const _Float16* gsrc,
                                                        unsigned ldsByteOff,
                                                        unsigned dim0Elems,   // row length
                                                        unsigned dim1Rows) {  // #rows
    unsigned long long ga = (unsigned long long)(size_t)gsrc;
    unsigned w0 = 1u;                                   // count=1, user mode
    unsigned w1 = ldsByteOff;                           // lds_addr
    unsigned w2 = (unsigned)ga;                         // global_addr[31:0]
    unsigned w3 = (unsigned)(ga >> 32) | (2u << 30);    // global_addr[56:32] | type=2

    unsigned g1w0 = (1u << 16)      // data_size = 2B
                  | (1u << 20)      // pad_enable
                  | (4u << 22)      // pad_interval: 32 DWORDs (128B)
                  | (3u << 25);     // pad_amount:   4 DWORDs (16B) -> 144B stride
    unsigned g1w1 = (dim0Elems & 0xFFFFu) << 16;                          // tensor_dim0 lo
    unsigned g1w2 = ((dim0Elems >> 16) & 0xFFFFu) | ((dim1Rows & 0xFFFFu) << 16);
    unsigned g1w3 = ((dim1Rows >> 16) & 0xFFFFu) | (64u << 16);           // tile_dim0 = 64
    unsigned g1w4 = 64u;                                                  // tile_dim1 = 64
    unsigned g1w5 = dim0Elems;                                            // dim0_stride lo
    unsigned g1w6 = 0u;
    unsigned g1w7 = 0u;

    u32x4 g0;
    g0.x = (unsigned)__builtin_amdgcn_readfirstlane((int)w0);
    g0.y = (unsigned)__builtin_amdgcn_readfirstlane((int)w1);
    g0.z = (unsigned)__builtin_amdgcn_readfirstlane((int)w2);
    g0.w = (unsigned)__builtin_amdgcn_readfirstlane((int)w3);
    u32x8 g1;
    g1[0] = (unsigned)__builtin_amdgcn_readfirstlane((int)g1w0);
    g1[1] = (unsigned)__builtin_amdgcn_readfirstlane((int)g1w1);
    g1[2] = (unsigned)__builtin_amdgcn_readfirstlane((int)g1w2);
    g1[3] = (unsigned)__builtin_amdgcn_readfirstlane((int)g1w3);
    g1[4] = (unsigned)__builtin_amdgcn_readfirstlane((int)g1w4);
    g1[5] = (unsigned)__builtin_amdgcn_readfirstlane((int)g1w5);
    g1[6] = (unsigned)__builtin_amdgcn_readfirstlane((int)g1w6);
    g1[7] = (unsigned)__builtin_amdgcn_readfirstlane((int)g1w7);
    asm volatile("tensor_load_to_lds %0, %1" :: "s"(g0), "s"(g1) : "memory");
}

// ---------------------------------------------------------------------------
// Packing kernels
// ---------------------------------------------------------------------------
__global__ void zero_half_kernel(unsigned* p, int n_u32) {
    int i = blockIdx.x * blockDim.x + threadIdx.x;
    if (i < n_u32) p[i] = 0u;
}

// NCHW f32 -> padded NHWC f16 (8, 66, 66, 256)
__global__ void pack_features_kernel(const float* __restrict__ src,
                                     _Float16* __restrict__ dst) {
    int i = blockIdx.x * blockDim.x + threadIdx.x;
    if (i >= B_IMG * C_IN * 64 * 64) return;
    int x = i & 63;
    int y = (i >> 6) & 63;
    int c = (i >> 12) & 255;
    int b = i >> 20;
    size_t o = ((size_t)(b * 66 + (y + 1)) * 66 + (x + 1)) * C_IN + c;
    dst[o] = (_Float16)src[i];
}

// conv_w (256,256,3,3) f32 -> wp[n][k], k = (ky*3+kx)*256 + c
__global__ void pack_w3_kernel(const float* __restrict__ w, _Float16* __restrict__ wp) {
    int i = blockIdx.x * blockDim.x + threadIdx.x;
    if (i >= C_IN * K3) return;
    int n = i / K3;
    int k = i - n * K3;
    int pos = k >> 8;
    int c = k & 255;
    int ky = pos / 3, kx = pos - ky * 3;
    wp[i] = (_Float16)w[((size_t)(n * C_IN + c) * 3 + ky) * 3 + kx];
}

// obj_w (12,256), delta_w (48,256) -> wh[64][256] f16 (pad rows 60..63 = 0)
__global__ void pack_wh_kernel(const float* __restrict__ ow, const float* __restrict__ dw,
                               _Float16* __restrict__ wh) {
    int i = blockIdx.x * blockDim.x + threadIdx.x;
    if (i >= N_HEAD * C_IN) return;
    int n = i >> 8;
    int c = i & 255;
    float v = 0.f;
    if (n < 12)      v = ow[n * C_IN + c];
    else if (n < 60) v = dw[(n - 12) * C_IN + c];
    wh[i] = (_Float16)v;
}

// ---------------------------------------------------------------------------
// WMMA fragment helpers (documented 16-bit layout; two ds_load_b128 per frag)
// ---------------------------------------------------------------------------
__device__ __forceinline__ v16h load_frag(const _Float16* rowBase, int halfSel) {
    v16h f;
    const _Float16* p0 = rowBase + halfSel * 8;
    reinterpret_cast<uint4*>(&f)[0] = *reinterpret_cast<const uint4*>(p0);
    reinterpret_cast<uint4*>(&f)[1] = *reinterpret_cast<const uint4*>(p0 + 16);
    return f;
}

#define WMMA_F16(ACC, A, Bf) \
    ACC = __builtin_amdgcn_wmma_f32_16x16x32_f16(false, A, false, Bf, (short)0, ACC, false, false)

// ---------------------------------------------------------------------------
// conv3x3 implicit GEMM: M=32768, N=256, K=2304 (f16 in, f32 acc)
// grid (256, 4), block 256 (8 waves). WG tile: 128(M) x 64(N).
// Double-buffered LDS, K=64 per stage (36 stages), 8 WMMA per barrier.
// ---------------------------------------------------------------------------
__global__ __launch_bounds__(256)
void conv3_gemm_kernel(const _Float16* __restrict__ feat,   // (8,66,66,256) padded NHWC
                       const _Float16* __restrict__ wp,     // (256, 2304)
                       const float*    __restrict__ bias,   // (256)
                       _Float16*       __restrict__ t)      // (32768, 256) NHWC
{
    __shared__ __align__(16) _Float16 As[2][128 * LDSH];
    __shared__ __align__(16) _Float16 Bs[2][64 * LDSH];

    const int tid  = threadIdx.x;
    const int wave = tid >> 5;
    const int lane = tid & 31;
    const int mBase = blockIdx.x * 128;
    const int nBase = blockIdx.y * 64;
    const int b  = mBase >> 12;
    const int y0 = (mBase & 4095) >> 6;   // tile = 2 full rows of x

    v8f acc0 = {}, acc1 = {}, acc2 = {}, acc3 = {};
    const int fr = lane & 15;
    const int fh = (lane >> 4) & 1;

    const unsigned ldsA[2] = { lds_off(&As[0][0]), lds_off(&As[1][0]) };
    const unsigned ldsB[2] = { lds_off(&Bs[0][0]), lds_off(&Bs[1][0]) };

    // Per-thread invariants: 4 chunks of 16B each (128 rows x 128B / 256 thr).
    unsigned ldsOffA[4], voffA[4];
    #pragma unroll
    for (int q = 0; q < 4; ++q) {
        int chunk = tid + q * 256;          // 0..1023
        int row = chunk >> 3;               // 0..127
        int sub = chunk & 7;                // 16B piece within 128B row
        int dy = row >> 6, x = row & 63;
        ldsOffA[q] = (unsigned)(row * 144 + sub * 16);
        voffA[q]   = (unsigned)(((dy * 66 + x) * C_IN) * 2 + sub * 16);
    }
    const unsigned long long featBase =
        (unsigned long long)(size_t)(feat + ((size_t)(b * 66 + y0) * 66) * C_IN);

    const int NK = 36;   // 9 taps * 4 c-slabs of 64
    int ky = 0, kx = 0, c0 = 0;   // indices of the NEXT stage to issue

    auto issue_stage = [&](int buf) {
        const unsigned long long sa =
            featBase + (unsigned long long)(((ky * 66 + kx) * C_IN + c0) * 2);
        #pragma unroll
        for (int q = 0; q < 4; ++q)
            async_g2l_b128_s(ldsA[buf] + ldsOffA[q], voffA[q], sa);
        if (wave == 0) {
            const _Float16* g = wp + (size_t)nBase * K3 + (ky * 3 + kx) * 256 + c0;
            tdm_load_tile_64x64_f16(g, ldsB[buf], K3, C_IN);
        }
    };
    auto advance = [&]() {
        c0 += 64;
        if (c0 == 256) { c0 = 0; if (++kx == 3) { kx = 0; ++ky; } }
    };

    issue_stage(0);
    advance();
    #pragma unroll 2
    for (int s = 0; s < NK; ++s) {
        const int buf = s & 1;
        wait_async0();
        if (wave == 0) __builtin_amdgcn_s_wait_tensorcnt(0);
        __syncthreads();
        if (s + 1 < NK) { issue_stage(buf ^ 1); advance(); }

        const _Float16* aP = &As[buf][(wave * 16 + fr) * LDSH];
        const _Float16* bP = &Bs[buf][fr * LDSH];
        #pragma unroll
        for (int ss = 0; ss < 2; ++ss) {
            v16h a  = load_frag(aP + ss * 32, fh);
            v16h b0 = load_frag(bP + ss * 32, fh);
            v16h b1 = load_frag(bP + 16 * LDSH + ss * 32, fh);
            v16h b2 = load_frag(bP + 32 * LDSH + ss * 32, fh);
            v16h b3 = load_frag(bP + 48 * LDSH + ss * 32, fh);
            WMMA_F16(acc0, a, b0);
            WMMA_F16(acc1, a, b1);
            WMMA_F16(acc2, a, b2);
            WMMA_F16(acc3, a, b3);
        }
    }

    // epilogue: bias + ReLU -> t (f16). C/D layout: VGPR j -> M=j (lanes 0-15), M=j+8 (16-31)
    const int nc = nBase + (lane & 15);
    const float bb0 = bias[nc], bb1 = bias[nc + 16], bb2 = bias[nc + 32], bb3 = bias[nc + 48];
    const int mw = mBase + wave * 16;
    #pragma unroll
    for (int j = 0; j < 8; ++j) {
        int mrow = mw + ((lane < 16) ? j : j + 8);
        _Float16* dst = t + (size_t)mrow * C_IN + nc;
        dst[0]  = (_Float16)fmaxf(acc0[j] + bb0, 0.f);
        dst[16] = (_Float16)fmaxf(acc1[j] + bb1, 0.f);
        dst[32] = (_Float16)fmaxf(acc2[j] + bb2, 0.f);
        dst[48] = (_Float16)fmaxf(acc3[j] + bb3, 0.f);
    }
}

// ---------------------------------------------------------------------------
// 1x1 heads GEMM: M=32768, N=64, K=256. grid (256), block 256. 4 stages.
// ---------------------------------------------------------------------------
__global__ __launch_bounds__(256)
void head_gemm_kernel(const _Float16* __restrict__ t,    // (32768,256)
                      const _Float16* __restrict__ wh,   // (64,256)
                      const float* __restrict__ obj_b,   // (12)
                      const float* __restrict__ del_b,   // (48)
                      float* __restrict__ obj,           // (32768*12)
                      float* __restrict__ deltas)        // (32768*12*4)
{
    __shared__ __align__(16) _Float16 As[2][128 * LDSH];
    __shared__ __align__(16) _Float16 Bs[2][64 * LDSH];

    const int tid  = threadIdx.x;
    const int wave = tid >> 5;
    const int lane = tid & 31;
    const int mBase = blockIdx.x * 128;

    v8f acc0 = {}, acc1 = {}, acc2 = {}, acc3 = {};
    const int fr = lane & 15;
    const int fh = (lane >> 4) & 1;

    const unsigned ldsA[2] = { lds_off(&As[0][0]), lds_off(&As[1][0]) };
    const unsigned ldsB[2] = { lds_off(&Bs[0][0]), lds_off(&Bs[1][0]) };

    unsigned ldsOffA[4], voffA[4];
    #pragma unroll
    for (int q = 0; q < 4; ++q) {
        int chunk = tid + q * 256;
        int row = chunk >> 3;
        int sub = chunk & 7;
        ldsOffA[q] = (unsigned)(row * 144 + sub * 16);
        voffA[q]   = (unsigned)((row * C_IN) * 2 + sub * 16);
    }
    const unsigned long long tBase =
        (unsigned long long)(size_t)(t + (size_t)mBase * C_IN);

    const int NK = 4;
    int c0 = 0;   // next stage to issue

    auto issue_stage = [&](int buf) {
        const unsigned long long sa = tBase + (unsigned long long)(c0 * 2);
        #pragma unroll
        for (int q = 0; q < 4; ++q)
            async_g2l_b128_s(ldsA[buf] + ldsOffA[q], voffA[q], sa);
        if (wave == 0) tdm_load_tile_64x64_f16(wh + c0, ldsB[buf], C_IN, N_HEAD);
    };

    issue_stage(0);
    c0 += 64;
    #pragma unroll 2
    for (int s = 0; s < NK; ++s) {
        const int buf = s & 1;
        wait_async0();
        if (wave == 0) __builtin_amdgcn_s_wait_tensorcnt(0);
        __syncthreads();
        if (s + 1 < NK) { issue_stage(buf ^ 1); c0 += 64; }

        const _Float16* aP = &As[buf][(wave * 16 + fr) * LDSH];
        const _Float16* bP = &Bs[buf][fr * LDSH];
        #pragma unroll
        for (int ss = 0; ss < 2; ++ss) {
            v16h a  = load_frag(aP + ss * 32, fh);
            v16h b0 = load_frag(bP + ss * 32, fh);
            v16h b1 = load_frag(bP + 16 * LDSH + ss * 32, fh);
            v16h b2 = load_frag(bP + 32 * LDSH + ss * 32, fh);
            v16h b3 = load_frag(bP + 48 * LDSH + ss * 32, fh);
            WMMA_F16(acc0, a, b0);
            WMMA_F16(acc1, a, b1);
            WMMA_F16(acc2, a, b2);
            WMMA_F16(acc3, a, b3);
        }
    }

    const int mw = mBase + wave * 16;
    #define HEAD_EPI(ACCV, NT)                                                    \
    {                                                                             \
        int n = (NT) * 16 + (lane & 15);                                          \
        float bb = (n < 12) ? obj_b[n] : ((n < 60) ? del_b[n - 12] : 0.f);        \
        _Pragma("unroll")                                                         \
        for (int j = 0; j < 8; ++j) {                                             \
            int mrow = mw + ((lane < 16) ? j : j + 8);                            \
            float v = ACCV[j] + bb;                                               \
            if (n < 12) obj[(size_t)mrow * ANCH + n] = v;                         \
            else if (n < 60) {                                                    \
                int d = n - 12;                                                   \
                deltas[((size_t)mrow * ANCH + (d >> 2)) * 4 + (d & 3)] = v;       \
            }                                                                     \
        }                                                                         \
    }
    HEAD_EPI(acc0, 0)
    HEAD_EPI(acc1, 1)
    HEAD_EPI(acc2, 2)
    HEAD_EPI(acc3, 3)
    #undef HEAD_EPI
}

// ---------------------------------------------------------------------------
// Sigmoid + anchor decode. One thread per anchor (8*49152).
// ---------------------------------------------------------------------------
__global__ void decode_kernel(const float* __restrict__ obj,
                              const float* __restrict__ deltas,
                              float* __restrict__ scores,
                              float* __restrict__ boxes) {
    int i = blockIdx.x * blockDim.x + threadIdx.x;
    if (i >= B_IMG * N_ANCH) return;
    scores[i] = 1.f / (1.f + __expf(-obj[i]));

    int a = i % ANCH;
    int cell = (i / ANCH) & (HW - 1);
    int x = cell & 63, y = cell >> 6;
    const float SC[4] = {32.f, 64.f, 128.f, 256.f};
    const float RT[3] = {0.5f, 1.f, 2.f};
    float s = SC[a / 3], r = RT[a % 3];
    float sq = sqrtf(r);
    float aw = s * sq, ah = s / sq;
    float cx = x * 16.f + 8.f, cy = y * 16.f + 8.f;

    float d0 = fminf(fmaxf(deltas[i * 4 + 0], -1.f), 1.f);
    float d1 = fminf(fmaxf(deltas[i * 4 + 1], -1.f), 1.f);
    float d2 = fminf(fmaxf(deltas[i * 4 + 2], -1.f), 1.f);
    float d3 = fminf(fmaxf(deltas[i * 4 + 3], -1.f), 1.f);
    float pcx = cx + d0 * aw;
    float pcy = cy + d1 * ah;
    float pw = fminf(fmaxf(aw * __expf(d2), 4.f), 1024.f);
    float ph = fminf(fmaxf(ah * __expf(d3), 4.f), 1024.f);
    float x1 = fminf(fmaxf(pcx - 0.5f * pw, 0.f), 1023.f);
    float y1 = fminf(fmaxf(pcy - 0.5f * ph, 0.f), 1023.f);
    float x2 = fminf(fmaxf(pcx + 0.5f * pw, 0.f), 1024.f);
    float y2 = fminf(fmaxf(pcy + 0.5f * ph, 0.f), 1024.f);
    reinterpret_cast<float4*>(boxes)[i] = make_float4(x1, y1, x2 - x1, y2 - y1);
}

// ---------------------------------------------------------------------------
// Top-1000 per image: radix threshold + atomic compaction. 1024 thr/WG.
// ---------------------------------------------------------------------------
__global__ __launch_bounds__(1024)
void topk_kernel(const float* __restrict__ scores,
                 const float* __restrict__ boxes,
                 float* __restrict__ topScores,   // (8*1000)
                 float* __restrict__ topBoxes)    // (8*1000*4)
{
    __shared__ unsigned hist[256];
    __shared__ unsigned sh_pref, sh_need, cntG, cntA, cntB;
    const int b = blockIdx.x;
    const int tid = threadIdx.x;
    const float* sc = scores + (size_t)b * N_ANCH;

    unsigned pref = 0, need = TOPK;
    for (int shift = 24; shift >= 0; shift -= 8) {
        if (tid < 256) hist[tid] = 0;
        __syncthreads();
        for (int i = tid; i < N_ANCH; i += 1024) {
            unsigned k = __float_as_uint(sc[i]);
            bool match = (shift == 24) || ((k >> (shift + 8)) == pref);
            if (match) atomicAdd(&hist[(k >> shift) & 255], 1u);
        }
        __syncthreads();
        if (tid == 0) {
            unsigned cum = 0;
            int bsel = 0;
            for (int bb = 255; bb >= 0; --bb) {
                unsigned h = hist[bb];
                if (cum + h >= need) { bsel = bb; break; }
                cum += h;
            }
            sh_pref = (pref << 8) | (unsigned)bsel;
            sh_need = need - cum;
        }
        __syncthreads();
        pref = sh_pref;
        need = sh_need;
        __syncthreads();
    }
    const unsigned T = pref;

    if (tid == 0) { cntG = 0; cntA = 0; cntB = 0; }
    __syncthreads();
    unsigned localG = 0;
    for (int i = tid; i < N_ANCH; i += 1024)
        if (__float_as_uint(sc[i]) > T) localG++;
    if (localG) atomicAdd(&cntG, localG);
    __syncthreads();
    const unsigned G = cntG;

    for (int i = tid; i < N_ANCH; i += 1024) {
        unsigned k = __float_as_uint(sc[i]);
        int slot = -1;
        if (k > T) {
            slot = (int)atomicAdd(&cntA, 1u);
        } else if (k == T) {
            unsigned e = atomicAdd(&cntB, 1u);
            if (G + e < TOPK) slot = (int)(G + e);
        }
        if (slot >= 0 && slot < TOPK) {
            topScores[b * TOPK + slot] = sc[i];
            reinterpret_cast<float4*>(topBoxes)[b * TOPK + slot] =
                reinterpret_cast<const float4*>(boxes)[(size_t)b * N_ANCH + i];
        }
    }
}

// ---------------------------------------------------------------------------
// Greedy NMS, 300 fixed steps. One 1024-thread WG per image, state in LDS.
// ---------------------------------------------------------------------------
__global__ __launch_bounds__(1024)
void nms_kernel(const float* __restrict__ topBoxes,
                const float* __restrict__ topScores,
                float* __restrict__ outBoxes,    // (8*300*4)
                float* __restrict__ outScores)   // (8*300)
{
    __shared__ float bx[TOPK], by[TOPK], bw[TOPK], bh[TOPK], ls[TOPK];
    __shared__ float redV[1024];
    __shared__ int   redI[1024];
    __shared__ float selB[4];
    const int b = blockIdx.x;
    const int tid = threadIdx.x;
    const float NEGINF = __uint_as_float(0xff800000u);

    if (tid < TOPK) {
        float4 bb = reinterpret_cast<const float4*>(topBoxes)[b * TOPK + tid];
        bx[tid] = bb.x; by[tid] = bb.y; bw[tid] = bb.z; bh[tid] = bb.w;
        ls[tid] = topScores[b * TOPK + tid];
    }
    __syncthreads();

    for (int it = 0; it < KEEP; ++it) {
        redV[tid] = (tid < TOPK) ? ls[tid] : NEGINF;
        redI[tid] = tid;
        __syncthreads();
        #pragma unroll
        for (int s2 = 512; s2 > 0; s2 >>= 1) {
            if (tid < s2 && redV[tid + s2] > redV[tid]) {
                redV[tid] = redV[tid + s2];
                redI[tid] = redI[tid + s2];
            }
            __syncthreads();
        }
        if (tid == 0) {
            int idx = redI[0];
            float s = redV[0];
            bool valid = s > -1e30f;
            float X = bx[idx], Y = by[idx], W = bw[idx], H = bh[idx];
            outScores[b * KEEP + it] = valid ? s : 0.f;
            float4 ob = valid ? make_float4(X, Y, W, H) : make_float4(0.f, 0.f, 0.f, 0.f);
            reinterpret_cast<float4*>(outBoxes)[b * KEEP + it] = ob;
            selB[0] = X; selB[1] = Y; selB[2] = X + W; selB[3] = Y + H;
            ls[idx] = NEGINF;
        }
        __syncthreads();
        if (tid < TOPK) {
            float x1 = bx[tid], y1 = by[tid];
            float x2 = x1 + bw[tid], y2 = y1 + bh[tid];
            float ix1 = fmaxf(x1, selB[0]), iy1 = fmaxf(y1, selB[1]);
            float ix2 = fminf(x2, selB[2]), iy2 = fminf(y2, selB[3]);
            float inter = fmaxf(ix2 - ix1, 0.f) * fmaxf(iy2 - iy1, 0.f);
            float a1 = fmaxf(x2 - x1, 0.f) * fmaxf(y2 - y1, 0.f);
            float a2 = fmaxf(selB[2] - selB[0], 0.f) * fmaxf(selB[3] - selB[1], 0.f);
            float uni = fmaxf(a1 + a2 - inter, 1e-6f);
            if (inter / uni > 0.7f) ls[tid] = NEGINF;
        }
        __syncthreads();
    }
}

// ---------------------------------------------------------------------------
// Launcher
// ---------------------------------------------------------------------------
extern "C" void kernel_launch(void* const* d_in, const int* in_sizes, int n_in,
                              void* d_out, int out_size, void* d_ws, size_t ws_size,
                              hipStream_t stream) {
    (void)in_sizes; (void)n_in; (void)out_size; (void)ws_size;
    const float* features = (const float*)d_in[0];
    const float* conv_w   = (const float*)d_in[1];
    const float* conv_b   = (const float*)d_in[2];
    const float* obj_w    = (const float*)d_in[3];
    const float* obj_b    = (const float*)d_in[4];
    const float* delta_w  = (const float*)d_in[5];
    const float* delta_b  = (const float*)d_in[6];
    // d_in[7] = image_size (1024) -- specialized to the reference shapes.

    char* ws = (char*)d_ws;
    size_t off = 0;
    auto carve = [&](size_t bytes) -> char* {
        off = (off + 255) & ~(size_t)255;
        char* p = ws + off;
        off += bytes;
        return p;
    };
    _Float16* feat_pad = (_Float16*)carve((size_t)B_IMG * 66 * 66 * C_IN * 2);
    _Float16* wp3      = (_Float16*)carve((size_t)C_IN * K3 * 2);
    _Float16* t_nhwc   = (_Float16*)carve((size_t)M_TOT * C_IN * 2);
    _Float16* wh       = (_Float16*)carve((size_t)N_HEAD * C_IN * 2);
    float* obj_buf     = (float*)carve((size_t)M_TOT * ANCH * 4);
    float* del_buf     = (float*)carve((size_t)M_TOT * ANCH * 4 * 4);
    float* sc_buf      = (float*)carve((size_t)B_IMG * N_ANCH * 4);
    float* bx_buf      = (float*)carve((size_t)B_IMG * N_ANCH * 4 * 4);
    float* topS        = (float*)carve((size_t)B_IMG * TOPK * 4);
    float* topB        = (float*)carve((size_t)B_IMG * TOPK * 4 * 4);

    float* outBoxes  = (float*)d_out;                      // (8,300,4)
    float* outScores = (float*)d_out + B_IMG * KEEP * 4;   // (8,300)

    {
        int n_u32 = (B_IMG * 66 * 66 * C_IN) / 2;
        zero_half_kernel<<<(n_u32 + 255) / 256, 256, 0, stream>>>((unsigned*)feat_pad, n_u32);
        int n = B_IMG * C_IN * 64 * 64;
        pack_features_kernel<<<(n + 255) / 256, 256, 0, stream>>>(features, feat_pad);
    }
    {
        int n3 = C_IN * K3;
        pack_w3_kernel<<<(n3 + 255) / 256, 256, 0, stream>>>(conv_w, wp3);
        int nh = N_HEAD * C_IN;
        pack_wh_kernel<<<(nh + 255) / 256, 256, 0, stream>>>(obj_w, delta_w, wh);
    }
    conv3_gemm_kernel<<<dim3(M_TOT / 128, C_IN / 64), 256, 0, stream>>>(
        feat_pad, wp3, conv_b, t_nhwc);
    head_gemm_kernel<<<M_TOT / 128, 256, 0, stream>>>(
        t_nhwc, wh, obj_b, delta_b, obj_buf, del_buf);
    {
        int n = B_IMG * N_ANCH;
        decode_kernel<<<(n + 255) / 256, 256, 0, stream>>>(obj_buf, del_buf, sc_buf, bx_buf);
    }
    topk_kernel<<<B_IMG, 1024, 0, stream>>>(sc_buf, bx_buf, topS, topB);
    nms_kernel<<<B_IMG, 1024, 0, stream>>>(topB, topS, outBoxes, outScores);
}
